// LorentzMultiHeadAttention_23381801959886
// MI455X (gfx1250) — compile-verified, matrix operand order checked by reading, not run on hardware
//
#include <hip/hip_runtime.h>
#include <hip/hip_bf16.h>

// ---------------------------------------------------------------------------
// LorentzMultiHeadAttention forward for gfx1250 (MI455X, wave32, WMMA)
//   B=2, L=2048, D=1024, H=16, d_h=64, N_T=4 time heads, N_S=12 space heads
// bf16 inputs + v_wmma_f32_16x16x32_bf16 (fp32 accum), fp32 softmax.
// GEMMs: double-buffered LDS staging via global_load_async_to_lds_b128.
// ---------------------------------------------------------------------------

typedef __bf16 bf16;
typedef __attribute__((ext_vector_type(16))) __bf16 v16bf;
typedef __attribute__((ext_vector_type(8)))  float  v8f;
typedef __attribute__((ext_vector_type(4)))  unsigned int u32x4;
typedef int v4i __attribute__((vector_size(16)));          // int4 vector
typedef __attribute__((address_space(1))) v4i* gv4p;       // global int4*
typedef __attribute__((address_space(3))) v4i* lv4p;       // LDS int4*

#define D_MODEL   1024
#define SEQ_LEN   2048
#define BATCH     2
#define N_HEADS   16
#define HEAD_DIM  64
#define QKV_COLS  3072   // 256 Qt | 256 Kt | 768 Qs | 768 Ks | 1024 V
#define TOKENS    (BATCH * SEQ_LEN)   // 4096

// ---------------------------------------------------------------------------
// 16-byte global -> LDS copy (async on CDNA5), plus completion wait.
// ---------------------------------------------------------------------------
__device__ inline void cp16(const bf16* __restrict__ g, bf16* l) {
#if __has_builtin(__builtin_amdgcn_global_load_async_to_lds_b128)
  __builtin_amdgcn_global_load_async_to_lds_b128(
      (gv4p)const_cast<bf16*>(g), (lv4p)l, 0, 0);
#else
  *(u32x4*)l = *(const u32x4*)g;
#endif
}

__device__ inline void cp_wait() {
#if __has_builtin(__builtin_amdgcn_global_load_async_to_lds_b128)
  #if __has_builtin(__builtin_amdgcn_s_wait_asynccnt)
    __builtin_amdgcn_s_wait_asynccnt(0);
  #else
    asm volatile("s_wait_asynccnt 0x0" ::: "memory");
  #endif
#endif
}

// ---------------------------------------------------------------------------
// Operand loaders matching the CDNA5 16-bit WMMA lane layout:
//   lanes 0-15  hold K = {0..7, 16..23}; lanes 16-31 hold K = {8..15, 24..31}
// ---------------------------------------------------------------------------
__device__ inline v16bf load_tile_rowmajor(const bf16* __restrict__ p, int ld) {
  const int lane = threadIdx.x & 31;
  const int r    = lane & 15;
  const int kh   = (lane >> 4) << 3;
  const bf16* q  = p + (size_t)r * ld;
  v16bf v;
#pragma unroll
  for (int i = 0; i < 8; ++i) v[i]     = q[kh + i];
#pragma unroll
  for (int i = 0; i < 8; ++i) v[8 + i] = q[16 + kh + i];
  return v;
}

// B-tile where column n is strided (element (k,n) at p + k*ld + n).
__device__ inline v16bf load_tile_colstrided(const bf16* __restrict__ p, int ld) {
  const int lane = threadIdx.x & 31;
  const int n    = lane & 15;
  const int kh   = (lane >> 4) << 3;
  v16bf v;
#pragma unroll
  for (int i = 0; i < 8; ++i) v[i]     = p[(size_t)(kh + i) * ld + n];
#pragma unroll
  for (int i = 0; i < 8; ++i) v[8 + i] = p[(size_t)(16 + kh + i) * ld + n];
  return v;
}

__device__ inline v8f wmma_bf16(v16bf a, v16bf b, v8f c) {
  return __builtin_amdgcn_wmma_f32_16x16x32_bf16(
      false, a, false, b, (short)0, c, false, false);
}

// ---------------------------------------------------------------------------
// fp32 -> bf16 elementwise conversion
// ---------------------------------------------------------------------------
__global__ void cvt_f32_bf16(const float* __restrict__ s, bf16* __restrict__ d, int n) {
  int i = blockIdx.x * blockDim.x + threadIdx.x;
  if (i < n) d[i] = (bf16)s[i];
}

// ---------------------------------------------------------------------------
// C[M,N] = A[M,K] * W[N,K]^T, double-buffered LDS staging.
// Block = 256 threads = 8 waves; tile 64(M) x 128(N) x 64(K).
// LDS: A 2x(64x64) + B 2x(128x64) bf16 = 48 KB.
// Each wave owns a 16x64 strip -> 4 accumulators, 8 WMMAs per K-step.
// ---------------------------------------------------------------------------
#define BM 64
#define BN 128
#define BK 64

__global__ __launch_bounds__(256)
void gemm_bf16_k(const bf16* __restrict__ A, const bf16* __restrict__ W,
                 int K, int ldc, bf16* __restrict__ Cb, float* __restrict__ Cf) {
  __shared__ bf16 As[2][BM * BK];
  __shared__ bf16 Bs[2][BN * BK];

  const int m0   = blockIdx.y * BM;
  const int n0   = blockIdx.x * BN;
  const int tid  = threadIdx.x;
  const int wave = tid >> 5;
  const int lane = tid & 31;
  const int mBase = m0 + (wave & 3) * 16;
  const int nBase = n0 + (wave >> 2) * 64;

  auto stage = [&](int buf, int k0) {
#pragma unroll
    for (int p = 0; p < 2; ++p) {               // A: 64 rows x 128 B
      int c = tid + p * 256;
      int r = c >> 3, kc = (c & 7) * 8;
      cp16(A + (size_t)(m0 + r) * K + k0 + kc, &As[buf][r * BK + kc]);
    }
#pragma unroll
    for (int p = 0; p < 4; ++p) {               // B: 128 rows x 128 B
      int c = tid + p * 256;
      int r = c >> 3, kc = (c & 7) * 8;
      cp16(W + (size_t)(n0 + r) * K + k0 + kc, &Bs[buf][r * BK + kc]);
    }
  };

  v8f acc[4] = {};

  stage(0, 0);
  cp_wait();
  __syncthreads();

  const int nk = K / BK;
  for (int kb = 0; kb < nk; ++kb) {
    const int buf = kb & 1;
    if (kb + 1 < nk) stage(buf ^ 1, (kb + 1) * BK);   // stream next tiles
#pragma unroll
    for (int kk = 0; kk < BK; kk += 32) {
      // Load A + all four B fragments first so the ds_loads can clause
      // under one dscnt wait, then issue the 4 WMMAs back-to-back.
      v16bf a  = load_tile_rowmajor(&As[buf][((wave & 3) * 16) * BK + kk], BK);
      v16bf b0 = load_tile_rowmajor(&Bs[buf][((wave >> 2) * 64 +  0) * BK + kk], BK);
      v16bf b1 = load_tile_rowmajor(&Bs[buf][((wave >> 2) * 64 + 16) * BK + kk], BK);
      v16bf b2 = load_tile_rowmajor(&Bs[buf][((wave >> 2) * 64 + 32) * BK + kk], BK);
      v16bf b3 = load_tile_rowmajor(&Bs[buf][((wave >> 2) * 64 + 48) * BK + kk], BK);
      acc[0] = wmma_bf16(a, b0, acc[0]);
      acc[1] = wmma_bf16(a, b1, acc[1]);
      acc[2] = wmma_bf16(a, b2, acc[2]);
      acc[3] = wmma_bf16(a, b3, acc[3]);
    }
    cp_wait();
    __syncthreads();
  }

  const int ro = (lane >> 4) * 8;   // accumulator row offset (ISA C/D layout)
  const int cc = lane & 15;
#pragma unroll
  for (int j = 0; j < 4; ++j) {
#pragma unroll
    for (int i = 0; i < 8; ++i) {
      const size_t idx = (size_t)(mBase + i + ro) * ldc + (nBase + j * 16 + cc);
      if (Cb) Cb[idx] = (bf16)acc[j][i];
      else    Cf[idx] = acc[j][i];
    }
  }
}

// ---------------------------------------------------------------------------
// Attention: one block per (b, h, 16-row q tile). 256 threads = 8 waves.
// LDS: S[16][2048] fp32 (128 KB) + P[16][2048] bf16 (64 KB) + red[256] fp32.
// Phase 1: scores via WMMA QK^T (8 waves split 128 key tiles)
// Phase 2: fp32 softmax; fp32 probs -> d_out, bf16 probs -> LDS
// Phase 3: out_h = P @ V; all 8 waves (split-K halves, LDS reduction)
// ---------------------------------------------------------------------------
__global__ __launch_bounds__(256)
void attn_k(const bf16* __restrict__ Y, const float* __restrict__ mask,
            const float* __restrict__ w_sigma, float* __restrict__ probs,
            bf16* __restrict__ AO) {
  extern __shared__ char smem[];
  float* S   = (float*)smem;                                       // 16x2048 f32
  bf16*  P   = (bf16*)(smem + 16 * SEQ_LEN * 4);                   // 16x2048 bf16
  float* red = (float*)(smem + 16 * SEQ_LEN * 4 + 16 * SEQ_LEN * 2);

  const int qt = blockIdx.x;
  const int h  = blockIdx.y;
  const int b  = blockIdx.z;
  const int q0 = qt * 16;
  const int wave = threadIdx.x >> 5;
  const int lane = threadIdx.x & 31;

  const float sig = 1.0f / (1.0f + __expf(-w_sigma[0]));
  int qoff, koff;
  float fac;
  if (h < 4) { qoff = h * 64;              koff = 256 + h * 64;          fac = -sig * 0.125f; }
  else       { qoff = 512 + (h - 4) * 64;  koff = 1280 + (h - 4) * 64;   fac = 0.125f; }
  const int voff = 2048 + h * 64;

  const bf16* Yb = Y + (size_t)b * SEQ_LEN * QKV_COLS;

  // ---- Phase 1: scores --------------------------------------------------
  v16bf a0 = load_tile_rowmajor(Yb + (size_t)q0 * QKV_COLS + qoff,      QKV_COLS);
  v16bf a1 = load_tile_rowmajor(Yb + (size_t)q0 * QKV_COLS + qoff + 32, QKV_COLS);

  for (int jt = wave; jt < SEQ_LEN / 16; jt += 8) {
    const int k0 = jt * 16;
    v16bf b0 = load_tile_rowmajor(Yb + (size_t)k0 * QKV_COLS + koff,      QKV_COLS);
    v16bf b1 = load_tile_rowmajor(Yb + (size_t)k0 * QKV_COLS + koff + 32, QKV_COLS);
    v8f c = {};
    c = wmma_bf16(a0, b0, c);
    c = wmma_bf16(a1, b1, c);
    const int col  = k0 + (lane & 15);
    const float mv = mask[b * SEQ_LEN + col];
    const int ro   = (lane >> 4) * 8;
#pragma unroll
    for (int i = 0; i < 8; ++i)
      S[(i + ro) * SEQ_LEN + col] = c[i] * fac + mv;
  }
  __syncthreads();

  // ---- Phase 2: softmax -------------------------------------------------
  const int row = threadIdx.x >> 4;
  const int sub = threadIdx.x & 15;
  float m = -3.4e38f;
  for (int j = sub; j < SEQ_LEN; j += 16) m = fmaxf(m, S[row * SEQ_LEN + j]);
  red[row * 16 + sub] = m;
  __syncthreads();
  float mx = red[row * 16];
#pragma unroll
  for (int t = 1; t < 16; ++t) mx = fmaxf(mx, red[row * 16 + t]);
  __syncthreads();

  float sum = 0.0f;
  for (int j = sub; j < SEQ_LEN; j += 16) {
    float e = __expf(S[row * SEQ_LEN + j] - mx);
    S[row * SEQ_LEN + j] = e;
    sum += e;
  }
  red[row * 16 + sub] = sum;
  __syncthreads();
  float denom = 0.0f;
#pragma unroll
  for (int t = 0; t < 16; ++t) denom += red[row * 16 + t];
  const float inv = 1.0f / denom;

  const size_t pbase = (((size_t)(b * N_HEADS + h)) * SEQ_LEN + q0) * SEQ_LEN;
  for (int j = sub; j < SEQ_LEN; j += 16) {
    float p = S[row * SEQ_LEN + j] * inv;
    probs[pbase + (size_t)row * SEQ_LEN + j] = p;   // fp32 attn_probs output
    P[row * SEQ_LEN + j] = (bf16)p;                 // bf16 for P@V
  }
  __syncthreads();

  // ---- Phase 3: out_h = P @ V (all 8 waves, split-K + LDS reduce) -------
  {
    const int n0    = (wave & 3) * 16;
    const int khalf = wave >> 2;                    // 0 or 1
    v8f c = {};
    const int kbeg = khalf * (SEQ_LEN / 2);
    const int kend = kbeg + (SEQ_LEN / 2);
    for (int kt = kbeg; kt < kend; kt += 32) {
      __builtin_prefetch(Yb + (size_t)(kt + 64) * QKV_COLS + voff + n0, 0, 3);
      v16bf pa = load_tile_rowmajor(P + kt, SEQ_LEN);                       // LDS
      v16bf vb = load_tile_colstrided(Yb + (size_t)kt * QKV_COLS + voff + n0,
                                      QKV_COLS);
      c = wmma_bf16(pa, vb, c);
    }
    // partial reduction through LDS (reuse dead score buffer S)
    float* part = S;
    if (wave >= 4) {
#pragma unroll
      for (int i = 0; i < 8; ++i) part[(wave - 4) * 256 + lane * 8 + i] = c[i];
    }
    __syncthreads();
    if (wave < 4) {
#pragma unroll
      for (int i = 0; i < 8; ++i) c[i] += part[wave * 256 + lane * 8 + i];
      const int ro = (lane >> 4) * 8;
      const int cc = lane & 15;
#pragma unroll
      for (int i = 0; i < 8; ++i) {
        const size_t idx = ((size_t)(b * SEQ_LEN) + q0 + i + ro) * D_MODEL
                           + h * HEAD_DIM + n0 + cc;
        AO[idx] = (bf16)c[i];
      }
    }
  }
}

// ---------------------------------------------------------------------------
// Host-side orchestration
// ---------------------------------------------------------------------------
extern "C" void kernel_launch(void* const* d_in, const int* in_sizes, int n_in,
                              void* d_out, int out_size, void* d_ws, size_t ws_size,
                              hipStream_t stream) {
  (void)in_sizes; (void)n_in; (void)out_size; (void)ws_size;

  const float* x       = (const float*)d_in[0];
  const float* amask   = (const float*)d_in[1];
  const float* w_qt    = (const float*)d_in[2];
  const float* w_kt    = (const float*)d_in[3];
  const float* w_qs    = (const float*)d_in[4];
  const float* w_ks    = (const float*)d_in[5];
  const float* w_sigma = (const float*)d_in[6];
  const float* w_v     = (const float*)d_in[7];
  const float* w_o     = (const float*)d_in[8];

  // Workspace layout (bf16 elements)
  bf16* xb   = (bf16*)d_ws;                      // 4096x1024
  bf16* Wall = xb   + (size_t)TOKENS * D_MODEL;  // 3072x1024 fused QKV weights
  bf16* Wo   = Wall + (size_t)QKV_COLS * D_MODEL;
  bf16* Yy   = Wo   + (size_t)D_MODEL * D_MODEL; // 4096x3072 fused QKV acts
  bf16* AO   = Yy   + (size_t)TOKENS * QKV_COLS; // 4096x1024 attn output

  float* out   = (float*)d_out;                  // (B,L,D)
  float* probs = out + (size_t)TOKENS * D_MODEL; // (B,H,L,L)

  auto cvt = [&](const float* s, bf16* d, int n) {
    cvt_f32_bf16<<<(n + 255) / 256, 256, 0, stream>>>(s, d, n);
  };
  cvt(x,    xb,                     TOKENS * D_MODEL);
  cvt(w_qt, Wall + 0,               256  * D_MODEL);
  cvt(w_kt, Wall + 256  * D_MODEL,  256  * D_MODEL);
  cvt(w_qs, Wall + 512  * D_MODEL,  768  * D_MODEL);
  cvt(w_ks, Wall + 1280 * D_MODEL,  768  * D_MODEL);
  cvt(w_v,  Wall + 2048 * D_MODEL,  1024 * D_MODEL);
  cvt(w_o,  Wo,                     D_MODEL * D_MODEL);

  {  // Y = x @ Wall^T
    dim3 grid(QKV_COLS / BN, TOKENS / BM);
    gemm_bf16_k<<<grid, 256, 0, stream>>>(xb, Wall, D_MODEL, QKV_COLS, Yy, nullptr);
  }

  {  // attention (scores + softmax + P@V); probs written straight to d_out
    dim3 grid(SEQ_LEN / 16, N_HEADS, BATCH);
    size_t lds = 16 * SEQ_LEN * 4 + 16 * SEQ_LEN * 2 + 256 * 4; // 197,632 B
    attn_k<<<grid, 256, lds, stream>>>(Yy, amask, w_sigma, probs, AO);
  }

  {  // out = AO @ Wo^T
    dim3 grid(D_MODEL / BN, TOKENS / BM);
    gemm_bf16_k<<<grid, 256, 0, stream>>>(AO, Wo, D_MODEL, D_MODEL, nullptr, out);
  }
}
